// Colorizer_68616397521135
// MI455X (gfx1250) — matrix operand order repaired and checked by reading, not compiled, not run on hardware
//
#include <hip/hip_runtime.h>
#include <hip/hip_bf16.h>

typedef __attribute__((ext_vector_type(16))) _Float16 v16h;
typedef __attribute__((ext_vector_type(8)))  _Float16 v8h;
typedef __attribute__((ext_vector_type(8)))  float    v8f;
typedef __attribute__((ext_vector_type(4)))  unsigned int u32x4;
typedef __attribute__((ext_vector_type(8)))  unsigned int u32x8;

#define H    56
#define W    56
#define CH   64
#define NPIX (H*W)        // 3136
#define PP   13
#define RR   6
#define NN   (PP*PP)      // 169
#define NREF 3
#define NTOT (NREF*NN)    // 507
#define CQ   32
#define X2BASE (-18)
#define BW   96           // padded Gram panel width, covers x2 in [-18,77]
#define ASTR 72           // LDS row stride (halves) for transposed panels (144B = 9*16)

// ---------------- prep: f32->f16 feature copies + stride-4 subsample of quantized_r
__global__ void k_prep(const float* __restrict__ feats_r,
                       const float* __restrict__ feats_t,
                       const float* __restrict__ q_in,
                       _Float16* __restrict__ fr16,
                       _Float16* __restrict__ ft16,
                       float* __restrict__ qr)
{
    int idx = blockIdx.x * blockDim.x + threadIdx.x;
    int stride = gridDim.x * blockDim.x;
    for (int t = idx; t < NREF*CH*NPIX; t += stride)
        fr16[t] = (_Float16)feats_r[t];
    for (int t = idx; t < CH*NPIX; t += stride)
        ft16[t] = (_Float16)feats_t[t];
    for (int t = idx; t < NREF*CQ*NPIX; t += stride) {
        int p  = t % NPIX;
        int rc = t / NPIX;                       // ref*CQ + c
        int yy = p / W, xx = p % W;
        qr[t] = q_in[(size_t)rc*224*224 + (size_t)(yy*4)*224 + (xx*4)];
    }
}

// Issue a 2-D TDM load: tile (tile_d1 rows x tile_d0 elems) of 2-byte elements,
// global row stride = stride0 elements, zero-fill beyond tensor_d0/tensor_d1.
// D# packing per CDNA5 ISA.
__device__ __forceinline__ void tdm_load_2d_f16(unsigned int lds_byte_off,
                                                const void* gaddr,
                                                unsigned int tensor_d0, unsigned int tensor_d1,
                                                unsigned int tile_d0,  unsigned int tile_d1,
                                                unsigned long long stride0_elems,
                                                unsigned int pad_en,
                                                unsigned int pad_interval_code,
                                                unsigned int pad_amount_code)
{
    unsigned long long ga = (unsigned long long)(size_t)gaddr;
    u32x4 g0;
    g0[0] = 1u;                                            // count=1, user mode
    g0[1] = lds_byte_off;                                  // lds_addr
    g0[2] = (unsigned int)ga;                              // global_addr[31:0]
    g0[3] = (unsigned int)((ga >> 32) & 0x1FFFFFFu)        // global_addr[56:32]
          | (2u << 30);                                    // type = 2 (image)
    u32x8 g1;
    g1[0] = (1u << 16)                                     // data_size = 1 -> 2 bytes
          | (pad_en << 20) | (pad_interval_code << 22) | (pad_amount_code << 25);
    g1[1] = (tensor_d0 & 0xFFFFu) << 16;                   // tensor_dim0[15:0]
    g1[2] = (tensor_d0 >> 16) | ((tensor_d1 & 0xFFFFu) << 16);
    g1[3] = (tensor_d1 >> 16) | (tile_d0 << 16);           // tile_dim0
    g1[4] = tile_d1;                                       // tile_dim1 (tile_dim2 = 0)
    g1[5] = (unsigned int)stride0_elems;                   // tensor_dim0_stride[31:0]
    g1[6] = (unsigned int)(stride0_elems >> 32) & 0xFFFFu; // stride[47:32]; dim1_stride lo = 0
    g1[7] = 0u;                                            // tensor_dim1_stride hi = 0
    asm volatile("tensor_load_to_lds %0, %1" :: "s"(g0), "s"(g1) : "memory");
}

// ---------------- correlation via WMMA row-Gram
// block = (i=dy index, y=row, ref); 256 threads (8 waves); each wave: 3 of 24 tiles
__global__ __launch_bounds__(256)
void k_corr_wmma(const _Float16* __restrict__ ft16,
                 const _Float16* __restrict__ fr16,
                 float* __restrict__ corrs)
{
    __shared__ __align__(16) _Float16 Al[64][ASTR];    // [x][c]   transposed panel
    __shared__ __align__(16) _Float16 Bt[BW][ASTR];    // [x2'][c] transposed panel
    __shared__ __align__(16) float    Gl[64][BW];      // [x][x2'] Gram; arena reused as TDM stage

    // TDM staging buffers carved from the G arena (consumed before G is written)
    _Float16* Astage = reinterpret_cast<_Float16*>(&Gl[0][0]);   // [64][64]  8KB
    _Float16* Bstage = Astage + 64*64;                           // [64][96] 12KB

    const int i   = blockIdx.x;           // 0..12  (dy)
    const int y   = blockIdx.y;           // 0..55
    const int ref = blockIdx.z;           // 0..2
    const int dil = (ref == 0) ? 3 : 1;
    const int y2  = y + (i - RR) * dil;
    const int tid = threadIdx.x;
    const bool y2ok = (y2 >= 0 && y2 < H);

    const _Float16* fr = fr16 + (size_t)ref * CH * NPIX;

    // Zero-fill the B window columns TDM does not cover: xi in [0,18) and [82,96).
    // (TDM writes xi in [18,82): global x in [0,64) with x>=56 zero-filled by OOB.)
    if (y2ok) {
        for (int t = tid; t < 64 * 32; t += 256) {
            int c = t >> 5, k = t & 31;
            int xi = (k < 18) ? k : (82 + (k - 18));
            Bstage[c * BW + xi] = (_Float16)0.f;
        }
    } else {
        for (int t = tid; t < 64 * BW; t += 256)
            Bstage[t] = (_Float16)0.f;
    }

    if (tid < 32) {   // wave 0 issues the tensor DMAs
        unsigned int a_lds = (unsigned int)(size_t)(void*)Astage;
        tdm_load_2d_f16(a_lds, ft16 + (size_t)y * W,
                        /*tensor_d0=*/W, /*tensor_d1=*/CH,
                        /*tile_d0=*/64, /*tile_d1=*/64,
                        /*stride0=*/NPIX, /*pad*/0, 0, 0);
        if (y2ok) {
            unsigned int b_lds = (unsigned int)(size_t)(void*)(Bstage + 18);
            // rows of 64 halves (32 dwords -> interval code 4), then skip 16 dwords
            // (code 15) to land each c-row at stride 96 halves.
            tdm_load_2d_f16(b_lds, fr + (size_t)y2 * W,
                            /*tensor_d0=*/W, /*tensor_d1=*/CH,
                            /*tile_d0=*/64, /*tile_d1=*/64,
                            /*stride0=*/NPIX, /*pad*/1, 4, 15);
        }
        __builtin_amdgcn_s_wait_tensorcnt(0);
    }
    if (y + 1 < H && (tid & 63) == 0)
        __builtin_prefetch(&ft16[(size_t)(y + 1) * W], 0, 3);
    __syncthreads();

    // Transpose stages into fragment-friendly [pixel][channel] layout.
    for (int t = tid; t < 64 * 64; t += 256) {
        int x = t & 63, c = t >> 6;
        Al[x][c] = Astage[c * 64 + x];
    }
    for (int t = tid; t < BW * 64; t += 256) {
        int xi = t % BW, c = t / BW;
        Bt[xi][c] = Bstage[c * BW + xi];
    }
    __syncthreads();

    const int wv   = tid >> 5;
    const int lane = tid & 31;
    const int lrow = lane & 15;
    const int hiA  = (lane < 16) ? 0 : 8;    // A/C lane-half K/M offset
    const int hiB  = (lane < 16) ? 0 : 16;   // unused; B fragments use hiA (same k order as A)

    for (int tt = 0; tt < 3; ++tt) {
        int tile = wv * 3 + tt;
        int mt = tile / 6, nt = tile % 6;
        v8f acc = {};
        #pragma unroll
        for (int ks = 0; ks < 2; ++ks) {
            int kb = ks * 32;
            int arow = mt * 16 + lrow;
            v8h alo = *(const v8h*)&Al[arow][kb + hiA];
            v8h ahi = *(const v8h*)&Al[arow][kb + 16 + hiA];
            v16h a = __builtin_shufflevector(alo, ahi,
                        0,1,2,3,4,5,6,7,8,9,10,11,12,13,14,15);
            int bcol = nt * 16 + lrow;
            v8h blo = *(const v8h*)&Bt[bcol][kb + hiA];
            v8h bhi = *(const v8h*)&Bt[bcol][kb + 16 + hiA];
            v16h b = __builtin_shufflevector(blo, bhi,
                        0,1,2,3,4,5,6,7,8,9,10,11,12,13,14,15);
            acc = __builtin_amdgcn_wmma_f32_16x16x32_f16(
                      false, a, false, b, (short)0, acc, false, false);
        }
        #pragma unroll
        for (int r = 0; r < 8; ++r)
            Gl[mt*16 + r + hiA][nt*16 + lrow] = acc[r];
    }
    __syncthreads();

    // cc[(i,j),y,x] = G[x][x + (j-6)*dil - X2BASE]
    for (int t = tid; t < PP * W; t += 256) {
        int j = t / W, x = t % W;
        int xi = x + (j - RR) * dil - X2BASE;
        corrs[(size_t)(ref*NN + i*PP + j) * NPIX + y*W + x] = Gl[x][xi];
    }
}

// ---------------- softmax over ref0's 169 shifts -> expected offset field (×dil0)
__global__ void k_soft_offset(const float* __restrict__ corrs,
                              float* __restrict__ offx, float* __restrict__ offy)
{
    int p = blockIdx.x * blockDim.x + threadIdx.x;
    if (p >= NPIX) return;
    float m = -1e30f;
    for (int n = 0; n < NN; ++n) m = fmaxf(m, corrs[(size_t)n*NPIX + p]);
    float s = 0.f, sx = 0.f, sy = 0.f;
    for (int n = 0; n < NN; ++n) {
        float e = __expf(corrs[(size_t)n*NPIX + p] - m);
        s  += e;
        sy += e * (float)(n / PP - RR);
        sx += e * (float)(n % PP - RR);
    }
    const float dil0 = 3.f;
    offx[p] = sx / s * dil0;
    offy[p] = sy / s * dil0;
}

// ---------------- deformable re-correlation for ref0 (overwrites corrs[0..169))
__global__ void k_deform_corr(const float* __restrict__ fr0,   // f32 ref0 [64][56][56]
                              const float* __restrict__ ft,    // f32 [64][56][56]
                              const float* __restrict__ offx,
                              const float* __restrict__ offy,
                              float* __restrict__ corrs)
{
    int idx = blockIdx.x * blockDim.x + threadIdx.x;
    if (idx >= NN * NPIX) return;
    int p = idx % NPIX, n = idx / NPIX;
    int y = p / W, x = p % W;
    float sy = (float)y + (float)(n / PP - RR) + offy[p];
    float sx = (float)x + (float)(n % PP - RR) + offx[p];
    float y0f = floorf(sy), x0f = floorf(sx);
    int   y0 = (int)y0f, x0 = (int)x0f;
    float wy = sy - y0f, wx = sx - x0f;
    float w00 = (1.f-wy)*(1.f-wx), w01 = (1.f-wy)*wx;
    float w10 = wy*(1.f-wx),       w11 = wy*wx;
    bool v0y = (y0   >= 0 && y0   < H), v1y = (y0+1 >= 0 && y0+1 < H);
    bool v0x = (x0   >= 0 && x0   < W), v1x = (x0+1 >= 0 && x0+1 < W);
    int yc0 = min(max(y0,   0), H-1), yc1 = min(max(y0+1, 0), H-1);
    int xc0 = min(max(x0,   0), W-1), xc1 = min(max(x0+1, 0), W-1);
    float dot = 0.f;
    for (int c = 0; c < CH; ++c) {
        const float* fc = fr0 + (size_t)c * NPIX;
        float s = 0.f;
        if (v0y && v0x) s += w00 * fc[yc0*W + xc0];
        if (v0y && v1x) s += w01 * fc[yc0*W + xc1];
        if (v1y && v0x) s += w10 * fc[yc1*W + xc0];
        if (v1y && v1x) s += w11 * fc[yc1*W + xc1];
        dot += ft[(size_t)c*NPIX + p] * s;
    }
    corrs[(size_t)n*NPIX + p] = dot;
}

// ---------------- softmax over all 507 entries per pixel (in place)
__global__ void k_softmax_all(float* __restrict__ corrs)
{
    int p = blockIdx.x * blockDim.x + threadIdx.x;
    if (p >= NPIX) return;
    float m = -1e30f;
    for (int k = 0; k < NTOT; ++k) m = fmaxf(m, corrs[(size_t)k*NPIX + p]);
    float s = 0.f;
    for (int k = 0; k < NTOT; ++k) s += __expf(corrs[(size_t)k*NPIX + p] - m);
    float inv = 1.f / s;
    for (int k = 0; k < NTOT; ++k) {
        size_t o = (size_t)k*NPIX + p;
        corrs[o] = __expf(corrs[o] - m) * inv;
    }
}

// ---------------- weighted aggregation; one wave per pixel, lane = channel
__global__ __launch_bounds__(256)
void k_aggregate(const float* __restrict__ corrs, const float* __restrict__ qr,
                 const float* __restrict__ offx,  const float* __restrict__ offy,
                 float* __restrict__ out)
{
    int wid = (blockIdx.x * blockDim.x + threadIdx.x) >> 5;
    int c   = threadIdx.x & 31;
    if (wid >= NPIX) return;
    int p = wid;
    int y = p / W, x = p % W;
    float ox = offx[p], oy = offy[p];
    float acc = 0.f;
    for (int ref = 0; ref < NREF; ++ref) {
        const float* q = qr + (size_t)(ref*CQ + c) * NPIX;
        for (int n = 0; n < NN; ++n) {
            float w  = corrs[(size_t)(ref*NN + n)*NPIX + p];
            int   ky = n / PP - RR, kx = n % PP - RR;
            float s;
            if (ref == 0) {
                float sy = (float)y + (float)ky + oy;
                float sx = (float)x + (float)kx + ox;
                float y0f = floorf(sy), x0f = floorf(sx);
                int   y0 = (int)y0f, x0 = (int)x0f;
                float wy = sy - y0f, wx = sx - x0f;
                float v00=0.f, v01=0.f, v10=0.f, v11=0.f;
                if (y0   >= 0 && y0   < H && x0   >= 0 && x0   < W) v00 = q[y0*W + x0];
                if (y0   >= 0 && y0   < H && x0+1 >= 0 && x0+1 < W) v01 = q[y0*W + x0+1];
                if (y0+1 >= 0 && y0+1 < H && x0   >= 0 && x0   < W) v10 = q[(y0+1)*W + x0];
                if (y0+1 >= 0 && y0+1 < H && x0+1 >= 0 && x0+1 < W) v11 = q[(y0+1)*W + x0+1];
                s = (1.f-wy)*((1.f-wx)*v00 + wx*v01) + wy*((1.f-wx)*v10 + wx*v11);
            } else {
                int yy = y + ky, xx = x + kx;
                s = (yy >= 0 && yy < H && xx >= 0 && xx < W) ? q[yy*W + xx] : 0.f;
            }
            acc += w * s;
        }
    }
    out[(size_t)c*NPIX + p] = acc;
}

extern "C" void kernel_launch(void* const* d_in, const int* in_sizes, int n_in,
                              void* d_out, int out_size, void* d_ws, size_t ws_size,
                              hipStream_t stream)
{
    (void)in_sizes; (void)n_in; (void)out_size; (void)ws_size;
    const float* feats_r = (const float*)d_in[0];   // [3][64][56][56]
    const float* feats_t = (const float*)d_in[1];   // [64][56][56]
    const float* q_in    = (const float*)d_in[2];   // [3][32][224][224]
    // d_in[3]=ref_index [0,30,39], d_in[4]=current_ind 40  => nsearch=1, dil0=3 (fixed by setup)

    char* ws = (char*)d_ws;
    size_t off = 0;
    auto take = [&](size_t bytes) -> char* {
        char* ptr = ws + off;
        off += (bytes + 255) & ~(size_t)255;
        return ptr;
    };
    float*    corrs = (float*)   take((size_t)NTOT*NPIX*sizeof(float));      // 6.36 MB
    float*    offx  = (float*)   take((size_t)NPIX*sizeof(float));
    float*    offy  = (float*)   take((size_t)NPIX*sizeof(float));
    _Float16* ft16  = (_Float16*)take((size_t)CH*NPIX*sizeof(_Float16));
    _Float16* fr16  = (_Float16*)take((size_t)NREF*CH*NPIX*sizeof(_Float16));
    float*    qr    = (float*)   take((size_t)NREF*CQ*NPIX*sizeof(float));
    float*    out   = (float*)d_out;                                          // [32][56][56]

    // 1) dtype conversion + subsample
    k_prep<<<2352, 256, 0, stream>>>(feats_r, feats_t, q_in, fr16, ft16, qr);

    // 2) WMMA correlation (TDM-fed): grid (i, y, ref)
    k_corr_wmma<<<dim3(PP, H, NREF), 256, 0, stream>>>(ft16, fr16, corrs);

    // 3) softmax over ref0 patch -> offset field
    k_soft_offset<<<(NPIX + 255)/256, 256, 0, stream>>>(corrs, offx, offy);

    // 4) deformable re-correlation for ref0
    k_deform_corr<<<(NN*NPIX + 255)/256, 256, 0, stream>>>(feats_r, feats_t, offx, offy, corrs);

    // 5) softmax over all 507 per pixel
    k_softmax_all<<<(NPIX + 255)/256, 256, 0, stream>>>(corrs);

    // 6) aggregation into output
    k_aggregate<<<(NPIX*32 + 255)/256, 256, 0, stream>>>(corrs, qr, offx, offy, out);
}
